// HVNet_69870527972051
// MI455X (gfx1250) — matrix operand order repaired and checked by reading, not compiled, not run on hardware
//
#include <hip/hip_runtime.h>
#include <hip/hip_bf16.h>
#include <math.h>

// ---------------------------------------------------------------------------
// HVNet (PaiNN-style hetero MPNN) for MI455X / gfx1250.
// All GEMMs via v_wmma_f32_16x16x32_f16 (f16 inputs, f32 accumulate).
// Activations kept in fp32 + f16 mirror so WMMA A-fragments are pure
// contiguous f16 loads (no v_cvt in the inner loop).
// N=4096 nodes, E=65536 edges, H=256, NRBF=128, L=3, T=2, NG=8.
// ---------------------------------------------------------------------------

#define NN   4096
#define EE   65536
#define HH   256
#define H3   768
#define H2   512
#define NRBF 128
#define LL   3
#define TT   2
#define NG   8

typedef __attribute__((ext_vector_type(16))) _Float16 v16h;
typedef __attribute__((ext_vector_type(8)))  _Float16 v8h;
typedef __attribute__((ext_vector_type(8)))  float    v8f;

__device__ __forceinline__ float scaled_silu(float z) {
    return z / (1.0f + __expf(-z)) * (1.0f / 0.6f);
}

// Build the 16-bit A 16x32 fragment (contiguous k order) from a row-major f16 row:
//   elem e -> K = k0 + e + 16h  => two contiguous 8-half runs.
__device__ __forceinline__ v16h load_a_frag(const _Float16* __restrict__ arow,
                                            int k0, int h) {
    v8h lo = *(const v8h*)(arow + k0 + 16 * h);
    v8h hi = *(const v8h*)(arow + k0 + 16 * h + 8);
    v16h a;
    #pragma unroll
    for (int q = 0; q < 8; ++q) { a[q] = lo[q]; a[8 + q] = hi[q]; }
    return a;
}

// ---------------------------------------------------------------------------
// Weight convert+transpose: W [slices, K, Ncols] f32  ->  Wt [slices, Ncols, K] f16
// ---------------------------------------------------------------------------
__global__ void convert_wT(const float* __restrict__ W, _Float16* __restrict__ Wt,
                           int slices, int K, int Ncols) {
    size_t id = (size_t)blockIdx.x * blockDim.x + threadIdx.x;
    size_t per = (size_t)K * Ncols;
    size_t total = (size_t)slices * per;
    if (id >= total) return;
    size_t s = id / per;
    size_t r = id - s * per;
    int k = (int)(r / Ncols);
    int n = (int)(r - (size_t)k * Ncols);
    Wt[s * per + (size_t)n * K + k] = (_Float16)W[id];
}

// ---------------------------------------------------------------------------
// Edge geometry: dist clamp, unit vector, Gaussian RBF w/ poly envelope (p=5).
// grid = E blocks, 128 threads (one per RBF bin).
// ---------------------------------------------------------------------------
__global__ void edge_geom(const float* __restrict__ pos,
                          const int* __restrict__ ej, const int* __restrict__ ei,
                          _Float16* __restrict__ rbf16, float* __restrict__ evec) {
    int e = blockIdx.x;
    int k = threadIdx.x;
    int j = ej[e], i = ei[e];
    float dxv = pos[j * 3 + 0] - pos[i * 3 + 0];
    float dyv = pos[j * 3 + 1] - pos[i * 3 + 1];
    float dzv = pos[j * 3 + 2] - pos[i * 3 + 2];
    float dist = sqrtf(dxv * dxv + dyv * dyv + dzv * dzv);
    if (fabsf(dist) < 1e-6f) dist = 1e-6f;
    if (k < 3) evec[(size_t)e * 3 + k] = (k == 0 ? dxv : (k == 1 ? dyv : dzv)) / dist;
    float d = dist * (1.0f / 5.0f);
    float env = 0.0f;
    if (d < 1.0f) {
        float d5 = d * d; d5 = d5 * d5 * d;             // d^5
        env = 1.0f - 21.0f * d5 + 35.0f * d5 * d - 15.0f * d5 * d * d;
    }
    float offk = (float)k * (1.0f / 127.0f);
    float diff = d - offk;
    rbf16[(size_t)e * NRBF + k] = (_Float16)(env * __expf(-8064.5f * diff * diff));
}

// ---------------------------------------------------------------------------
// x init from embedding table (fp32 + f16 mirror).
// ---------------------------------------------------------------------------
__global__ void init_x(const float* __restrict__ emb, const int* __restrict__ ntype,
                       float* __restrict__ x, _Float16* __restrict__ x16) {
    int id = blockIdx.x * blockDim.x + threadIdx.x;   // N*H
    int n = id >> 8, c = id & 255;
    float v = emb[(size_t)ntype[n] * HH + c];
    x[id] = v;
    x16[id] = (_Float16)v;
}

// ---------------------------------------------------------------------------
// Generic WMMA GEMM:  C = act( A16[M,K] @ Bt[Ncols,K]^T + bias ), optional f16
// mirror of C. One wave = one 16x16 tile. Block = 8 waves (16 rows x 128 cols).
//   B fragment (16-bit B 32x16, contiguous k order as A): elem e -> K = k0 + e + 16h, col = lane&15
// ---------------------------------------------------------------------------
__global__ void gemm_wmma(const _Float16* __restrict__ A16, const _Float16* __restrict__ Bt,
                          const float* __restrict__ bias, float* __restrict__ C,
                          _Float16* __restrict__ C16,
                          int M, int Ncols, int K, int act) {
    const int lane = threadIdx.x & 31;
    const int wave = threadIdx.x >> 5;
    const int l15  = lane & 15;
    const int h    = (lane >> 4) & 1;
    const int m0   = blockIdx.x * 16;
    const int n0   = blockIdx.y * 128 + wave * 16;
    if (n0 >= Ncols || m0 >= M) return;

    const _Float16* arow = A16 + (size_t)(m0 + l15) * K;
    const _Float16* brow = Bt  + (size_t)(n0 + l15) * K;

    v8f acc = {0.f, 0.f, 0.f, 0.f, 0.f, 0.f, 0.f, 0.f};
    for (int k0 = 0; k0 < K; k0 += 32) {
        __builtin_prefetch(arow + k0 + 128, 0, 1);   // global_prefetch_b8
        v16h a = load_a_frag(arow, k0, h);
        v16h b = *(const v16h*)(brow + k0 + 16 * h);
        acc = __builtin_amdgcn_wmma_f32_16x16x32_f16(false, a, false, b,
                                                     (short)0, acc, false, false);
    }

    const int col = n0 + l15;
    const float bv = bias ? bias[col] : 0.0f;
    #pragma unroll
    for (int v = 0; v < 8; ++v) {
        int row = m0 + v + 8 * h;
        float val = acc[v] + bv;
        if (act) val = scaled_silu(val);
        C[(size_t)row * Ncols + col] = val;
        if (C16) C16[(size_t)row * Ncols + col] = (_Float16)val;
    }
}

// ---------------------------------------------------------------------------
// Fused edge kernel: rh = rbf @ Wr_t + br  (WMMA, K=128), m = xh[j]*rh,
// masked atomic scatter of mh1/mh2/mh3 into dv/dx. grid = (E/16, 6).
//   seg 0,1 -> mh1 (cols 0..255):   dv += vec[j]*m/sqrt(H)
//   seg 2,3 -> mh2 (cols 256..511): dx += m/sqrt(3)
//   seg 4,5 -> mh3 (cols 512..767): dv += evec[e]*m/sqrt(H)
// ---------------------------------------------------------------------------
__global__ void edge_msg_wmma(const _Float16* __restrict__ rbf16,   // [E,128]
                              const _Float16* __restrict__ Wrt,     // [768,128]
                              const float* __restrict__ br,         // [768]
                              const float* __restrict__ xh,         // [N,768]
                              const float* __restrict__ vec,        // [N,3,H]
                              const float* __restrict__ evec,       // [E,3]
                              const int* __restrict__ ej, const int* __restrict__ ei,
                              const int* __restrict__ ntype,
                              float* __restrict__ dx, float* __restrict__ dv, int t) {
    const int e0 = blockIdx.x * 16;
    // Early-out: skip block if no edge in the tile targets a type-t node.
    bool any = false;
    #pragma unroll
    for (int q = 0; q < 16; ++q)
        if (ntype[ei[e0 + q]] == t) any = true;
    if (!any) return;

    const int lane = threadIdx.x & 31;
    const int wave = threadIdx.x >> 5;
    const int l15  = lane & 15;
    const int h    = (lane >> 4) & 1;
    const int seg  = blockIdx.y;                 // 0..5
    const int n0   = seg * 128 + wave * 16;      // column base in [0,768)

    const _Float16* arow = rbf16 + (size_t)(e0 + l15) * NRBF;
    const _Float16* brow = Wrt   + (size_t)(n0 + l15) * NRBF;

    v8f acc = {0.f, 0.f, 0.f, 0.f, 0.f, 0.f, 0.f, 0.f};
    #pragma unroll
    for (int k0 = 0; k0 < NRBF; k0 += 32) {
        v16h a = load_a_frag(arow, k0, h);
        v16h b = *(const v16h*)(brow + k0 + 16 * h);
        acc = __builtin_amdgcn_wmma_f32_16x16x32_f16(false, a, false, b,
                                                     (short)0, acc, false, false);
    }

    const int   col = n0 + l15;
    const float bv  = br[col];
    const float inv_sqrt_h = 0.0625f;               // 1/sqrt(256)
    const float inv_sqrt_3 = 0.57735026918962576f;

    #pragma unroll
    for (int v = 0; v < 8; ++v) {
        int e = e0 + v + 8 * h;
        int i = ei[e];
        if (ntype[i] != t) continue;
        int j = ej[e];
        float rh = acc[v] + bv;
        float m  = xh[(size_t)j * H3 + col] * rh;
        if (seg < 2) {                               // mh1
            int c = col;
            float s = m * inv_sqrt_h;
            #pragma unroll
            for (int d = 0; d < 3; ++d)
                atomicAdd(&dv[((size_t)i * 3 + d) * HH + c],
                          vec[((size_t)j * 3 + d) * HH + c] * s);
        } else if (seg < 4) {                        // mh2
            int c = col - 256;
            atomicAdd(&dx[(size_t)i * HH + c], m * inv_sqrt_3);
        } else {                                     // mh3
            int c = col - 512;
            float s = m * inv_sqrt_h;
            #pragma unroll
            for (int d = 0; d < 3; ++d)
                atomicAdd(&dv[((size_t)i * 3 + d) * HH + c], evec[(size_t)e * 3 + d] * s);
        }
    }
}

// ---------------------------------------------------------------------------
// xt = x + dx ; vt = vec + dv (with f16 mirror of vt for the Wv GEMM).
// N*4*H threads.
// ---------------------------------------------------------------------------
__global__ void add_dx_dv(const float* __restrict__ x, const float* __restrict__ dx,
                          const float* __restrict__ vec, const float* __restrict__ dv,
                          float* __restrict__ xt, float* __restrict__ vt,
                          _Float16* __restrict__ vt16) {
    int id = blockIdx.x * blockDim.x + threadIdx.x;
    if (id < NN * HH) {
        xt[id] = x[id] + dx[id];
    } else {
        int o = id - NN * HH;
        float v = vec[o] + dv[o];
        vt[o] = v;
        vt16[o] = (_Float16)v;
    }
}

// ---------------------------------------------------------------------------
// From vp=[N,3,2H]: vdot = sum_d v1*v2/sqrt(H); cat = [xt, ||v2||] (f32+f16).
// N*H threads.
// ---------------------------------------------------------------------------
__global__ void update_mid(const float* __restrict__ vp, const float* __restrict__ xt,
                           float* __restrict__ vdot, _Float16* __restrict__ cat16) {
    int id = blockIdx.x * blockDim.x + threadIdx.x;
    int n = id >> 8, c = id & 255;
    float s = 0.f, q = 0.f;
    #pragma unroll
    for (int d = 0; d < 3; ++d) {
        float v1 = vp[((size_t)n * 3 + d) * H2 + c];
        float v2 = vp[((size_t)n * 3 + d) * H2 + HH + c];
        s += v1 * v2; q += v2 * v2;
    }
    vdot[id] = s * 0.0625f;
    cat16[(size_t)n * H2 + c]      = (_Float16)xt[id];
    cat16[(size_t)n * H2 + HH + c] = (_Float16)sqrtf(q + 1e-12f);
}

// ---------------------------------------------------------------------------
// Hetero-masked final combine for type t (fp32 + f16 mirror of x).
// N*H threads.
// ---------------------------------------------------------------------------
__global__ void update_out(const float* __restrict__ ah, const float* __restrict__ vdot,
                           const float* __restrict__ xt, const float* __restrict__ vt,
                           const float* __restrict__ vp, const int* __restrict__ ntype,
                           float* __restrict__ xnew, _Float16* __restrict__ x16new,
                           float* __restrict__ vnew, int t) {
    int id = blockIdx.x * blockDim.x + threadIdx.x;
    int n = id >> 8, c = id & 255;
    if (ntype[n] != t) return;
    float a1 = ah[(size_t)n * H3 + c];
    float a2 = ah[(size_t)n * H3 + HH + c];
    float a3 = ah[(size_t)n * H3 + 2 * HH + c];
    const float is2 = 0.70710678118654752f;
    float xv = xt[id] + (a2 + a3 * vdot[id]) * is2;
    xnew[id] = xv;
    x16new[id] = (_Float16)xv;
    #pragma unroll
    for (int d = 0; d < 3; ++d) {
        size_t o = ((size_t)n * 3 + d) * HH + c;
        vnew[o] = vt[o] + a1 * vp[((size_t)n * 3 + d) * H2 + c];
    }
}

// ---------------------------------------------------------------------------
// Energy head tail: e_atom = hh . oW2 + ob2 ; atomic segment-sum into energy.
// ---------------------------------------------------------------------------
__global__ void head_reduce(const float* __restrict__ hh, const float* __restrict__ oW2,
                            const float* __restrict__ ob2, const int* __restrict__ batch,
                            float* __restrict__ energy) {
    int n = blockIdx.x * blockDim.x + threadIdx.x;
    if (n >= NN) return;
    float s = 0.f;
    for (int k = 0; k < 128; ++k) s += hh[(size_t)n * 128 + k] * oW2[k];
    s += ob2[0];
    atomicAdd(&energy[batch[n]], s);
}

// ---------------------------------------------------------------------------
extern "C" void kernel_launch(void* const* d_in, const int* in_sizes, int n_in,
                              void* d_out, int out_size, void* d_ws, size_t ws_size,
                              hipStream_t stream) {
    (void)in_sizes; (void)n_in; (void)out_size; (void)ws_size;
    const float* pos    = (const float*)d_in[0];
    const int*   ntype  = (const int*)d_in[1];
    const int*   eidx   = (const int*)d_in[2];
    const int*   batch  = (const int*)d_in[3];
    const float* emb    = (const float*)d_in[4];
    const float* W1  = (const float*)d_in[5];
    const float* b1  = (const float*)d_in[6];
    const float* W2  = (const float*)d_in[7];
    const float* b2  = (const float*)d_in[8];
    const float* Wr  = (const float*)d_in[9];
    const float* br  = (const float*)d_in[10];
    const float* Wv  = (const float*)d_in[11];
    const float* Wx1 = (const float*)d_in[12];
    const float* bx1 = (const float*)d_in[13];
    const float* Wx2 = (const float*)d_in[14];
    const float* bx2 = (const float*)d_in[15];
    const float* oW1 = (const float*)d_in[16];
    const float* ob1 = (const float*)d_in[17];
    const float* oW2 = (const float*)d_in[18];
    const float* ob2 = (const float*)d_in[19];
    float* out = (float*)d_out;

    const int* ej = eidx;         // sources
    const int* ei = eidx + EE;    // destinations

    // ---- workspace carve-up (256B aligned) ----
    char* ws = (char*)d_ws;
    size_t off = 0;
    auto alloc = [&](size_t bytes) -> char* {
        char* p = ws + off;
        off = (off + bytes + 255) & ~(size_t)255;
        return p;
    };
    float* xA   = (float*)alloc((size_t)NN * HH * 4);
    float* xB   = (float*)alloc((size_t)NN * HH * 4);
    float* vecA = (float*)alloc((size_t)NN * 3 * HH * 4);
    float* vecB = (float*)alloc((size_t)NN * 3 * HH * 4);
    float* h1   = (float*)alloc((size_t)NN * HH * 4);        // also ah1 / head hidden
    float* xh   = (float*)alloc((size_t)NN * H3 * 4);        // also ah
    float* dxb  = (float*)alloc((size_t)NN * HH * 4);
    float* dvb  = (float*)alloc((size_t)NN * 3 * HH * 4);
    float* xt   = (float*)alloc((size_t)NN * HH * 4);
    float* vt   = (float*)alloc((size_t)NN * 3 * HH * 4);
    float* vp   = (float*)alloc((size_t)NN * 3 * H2 * 4);
    float* vdot = (float*)alloc((size_t)NN * HH * 4);
    float* evec = (float*)alloc((size_t)EE * 3 * 4);
    _Float16* x16A  = (_Float16*)alloc((size_t)NN * HH * 2);
    _Float16* x16B  = (_Float16*)alloc((size_t)NN * HH * 2);
    _Float16* h116  = (_Float16*)alloc((size_t)NN * HH * 2);
    _Float16* vt16  = (_Float16*)alloc((size_t)NN * 3 * HH * 2);
    _Float16* cat16 = (_Float16*)alloc((size_t)NN * H2 * 2);
    _Float16* rbf16 = (_Float16*)alloc((size_t)EE * NRBF * 2);
    _Float16* W1t  = (_Float16*)alloc((size_t)LL * TT * HH * HH * 2);
    _Float16* W2t  = (_Float16*)alloc((size_t)LL * TT * HH * H3 * 2);
    _Float16* Wrt  = (_Float16*)alloc((size_t)LL * TT * NRBF * H3 * 2);
    _Float16* Wvt  = (_Float16*)alloc((size_t)LL * TT * HH * H2 * 2);
    _Float16* Wx1t = (_Float16*)alloc((size_t)LL * TT * H2 * HH * 2);
    _Float16* Wx2t = (_Float16*)alloc((size_t)LL * TT * HH * H3 * 2);
    _Float16* oW1t = (_Float16*)alloc((size_t)HH * 128 * 2);

    auto cvt = [&](const float* W, _Float16* Wt, int slices, int K, int Nc) {
        size_t total = (size_t)slices * K * Nc;
        convert_wT<<<dim3((unsigned)((total + 255) / 256)), 256, 0, stream>>>(W, Wt, slices, K, Nc);
    };
    cvt(W1,  W1t,  LL * TT, HH,   HH);
    cvt(W2,  W2t,  LL * TT, HH,   H3);
    cvt(Wr,  Wrt,  LL * TT, NRBF, H3);
    cvt(Wv,  Wvt,  LL * TT, HH,   H2);
    cvt(Wx1, Wx1t, LL * TT, H2,   HH);
    cvt(Wx2, Wx2t, LL * TT, HH,   H3);
    cvt(oW1, oW1t, 1,       HH,   128);

    edge_geom<<<dim3(EE), 128, 0, stream>>>(pos, ej, ei, rbf16, evec);
    init_x<<<dim3(NN * HH / 256), 256, 0, stream>>>(emb, ntype, xA, x16A);
    hipMemsetAsync(vecA, 0, (size_t)NN * 3 * HH * 4, stream);

    float* xcur = xA;  float* xnew = xB;
    _Float16* x16cur = x16A; _Float16* x16new = x16B;
    float* vcur = vecA; float* vnew = vecB;

    for (int l = 0; l < LL; ++l) {
        for (int t = 0; t < TT; ++t) {
            size_t lt = (size_t)(l * TT + t);
            const _Float16* W1t_  = W1t  + lt * HH * HH;
            const _Float16* W2t_  = W2t  + lt * HH * H3;
            const _Float16* Wrt_  = Wrt  + lt * NRBF * H3;
            const _Float16* Wvt_  = Wvt  + lt * HH * H2;
            const _Float16* Wx1t_ = Wx1t + lt * H2 * HH;
            const _Float16* Wx2t_ = Wx2t + lt * HH * H3;
            const float* b1_  = b1  + lt * HH;
            const float* b2_  = b2  + lt * H3;
            const float* br_  = br  + lt * H3;
            const float* bx1_ = bx1 + lt * HH;
            const float* bx2_ = bx2 + lt * H3;

            // xh = scaled_silu(x@W1+b1) @ W2 + b2
            gemm_wmma<<<dim3(NN / 16, HH / 128), 256, 0, stream>>>(
                x16cur, W1t_, b1_, h1, h116, NN, HH, HH, 1);
            gemm_wmma<<<dim3(NN / 16, H3 / 128), 256, 0, stream>>>(
                h116, W2t_, b2_, xh, nullptr, NN, H3, HH, 0);

            // edge message block (rh GEMM fused with gather + atomic scatter)
            hipMemsetAsync(dxb, 0, (size_t)NN * HH * 4, stream);
            hipMemsetAsync(dvb, 0, (size_t)NN * 3 * HH * 4, stream);
            edge_msg_wmma<<<dim3(EE / 16, 6), 256, 0, stream>>>(
                rbf16, Wrt_, br_, xh, vcur, evec, ej, ei, ntype, dxb, dvb, t);

            // update block
            add_dx_dv<<<dim3(NN * 4 * HH / 256), 256, 0, stream>>>(
                xcur, dxb, vcur, dvb, xt, vt, vt16);
            gemm_wmma<<<dim3(3 * NN / 16, H2 / 128), 256, 0, stream>>>(
                vt16, Wvt_, nullptr, vp, nullptr, 3 * NN, H2, HH, 0);
            update_mid<<<dim3(NN * HH / 256), 256, 0, stream>>>(vp, xt, vdot, cat16);
            gemm_wmma<<<dim3(NN / 16, HH / 128), 256, 0, stream>>>(
                cat16, Wx1t_, bx1_, h1, h116, NN, HH, H2, 1);
            gemm_wmma<<<dim3(NN / 16, H3 / 128), 256, 0, stream>>>(
                h116, Wx2t_, bx2_, xh, nullptr, NN, H3, HH, 0);
            update_out<<<dim3(NN * HH / 256), 256, 0, stream>>>(
                xh, vdot, xt, vt, vp, ntype, xnew, x16new, vnew, t);
        }
        // hetero combine done for both types -> swap ping-pong buffers
        float* tmp;
        _Float16* tmp16;
        tmp = xcur; xcur = xnew; xnew = tmp;
        tmp16 = x16cur; x16cur = x16new; x16new = tmp16;
        tmp = vcur; vcur = vnew; vnew = tmp;
    }

    // energy head: hh = scaled_silu(x@oW1+ob1); e_atom = hh.oW2+ob2; segment-sum
    gemm_wmma<<<dim3(NN / 16, 1), 256, 0, stream>>>(
        x16cur, oW1t, ob1, h1, nullptr, NN, 128, HH, 1);
    hipMemsetAsync(out, 0, NG * sizeof(float), stream);
    head_reduce<<<dim3(NN / 256), 256, 0, stream>>>(h1, oW2, ob2, batch, out);
}
